// StarGATe_86466281603523
// MI455X (gfx1250) — matrix operand order repaired and checked by reading, not compiled, not run on hardware
//
#include <hip/hip_runtime.h>
#include <hip/hip_bf16.h>

typedef __attribute__((ext_vector_type(16))) _Float16 v16h;
typedef __attribute__((ext_vector_type(8)))  _Float16 v8h;
typedef __attribute__((ext_vector_type(8)))  float    v8f;

#define GN 20000
#define GE 320000
#define GH 4
#define GC 64
#define GD 256
#define GB 64
#define NEG_SLOPE 0.2f
#define BN_EPS 1e-5f

// ---------------------------------------------------------------------------
// monotone float <-> uint encoding for atomicMax on floats
// ---------------------------------------------------------------------------
__device__ __forceinline__ unsigned fenc(float f) {
    unsigned u = __float_as_uint(f);
    return (u & 0x80000000u) ? ~u : (u | 0x80000000u);
}
__device__ __forceinline__ float fdec(unsigned e) {
    unsigned u = (e & 0x80000000u) ? (e ^ 0x80000000u) : ~e;
    return __uint_as_float(u);
}

// ---------------------------------------------------------------------------
// f32 -> f16 convert with zero padding (rows and cols). dcols must be pow2.
// ---------------------------------------------------------------------------
__global__ void cvt_f16_pad(const float* __restrict__ src, _Float16* __restrict__ dst,
                            int srows, int scols, int drows, int dcols_lg2)
{
    size_t i = (size_t)blockIdx.x * blockDim.x + threadIdx.x;
    int dcols = 1 << dcols_lg2;
    if (i >= ((size_t)drows << dcols_lg2)) return;
    int r = (int)(i >> dcols_lg2);
    int c = (int)(i & (dcols - 1));
    float v = (r < srows && c < scols) ? src[(size_t)r * scols + c] : 0.0f;
    dst[i] = (_Float16)v;
}

// ---------------------------------------------------------------------------
// Tiled WMMA GEMM:  Y[M,Nw] = A[M,K](f16) * W[K,Nw](f16) (+bias)(+relu), f32 out
// Block = 256 threads = 8 waves; block tile 64(M) x 64(N); K stepped by 32.
// Wave grid 4x2; each wave: one A frag vs two B frags -> two 16x16 D tiles.
// K must be a multiple of 32 (operands are pre-padded). lda/ldw in elements.
// ---------------------------------------------------------------------------
__global__ __launch_bounds__(256) void wmma_gemm_f16(
    const _Float16* __restrict__ A, int lda,
    const _Float16* __restrict__ W, int ldw,
    const float* __restrict__ bias,
    float* __restrict__ Y, int ldy,
    int M, int K, int relu)
{
    __shared__ alignas(16) _Float16 sA [64][32];   // [m][k]  4 KB
    __shared__ alignas(16) _Float16 sBt[64][32];   // [n][k]  4 KB (transposed)

    const int tid  = threadIdx.x;
    const int lane = tid & 31;
    const int wave = tid >> 5;
    const int wm   = (wave & 3) << 4;      // wave row offset in tile
    const int wn0  = (wave >> 2) << 5;     // wave col offset (two 16-wide tiles)
    const int bm   = blockIdx.x * 64;
    const int bn   = blockIdx.y * 64;

    // staging coordinates
    const int ar = tid >> 2;               // A row 0..63
    const int ak = (tid & 3) << 3;         // A k-chunk (8 halves)
    const int wk = tid >> 3;               // W k-row 0..31
    const int wc = (tid & 7) << 3;         // W n-chunk (8 halves)
    const int gm = bm + ar;

    // fragment coordinates (ISA 7.12.2, wave32)
    const int mrow  = wm + (lane & 15);
    const int nidx  = lane & 15;
    const int ahalf = (lane >> 4) << 3;    // A: K += 8  for lanes 16..31
    const int bhalf = (lane >> 4) << 3;    // B: K += 8  for lanes 16..31

    v8f acc0 = {}, acc1 = {};
    const v8h zero8 = {};

    const int ksteps = K >> 5;
    for (int ks = 0; ks < ksteps; ++ks) {
        const int k0 = ks << 5;
        // stage A tile: one b128 load + one b128 LDS store per thread
        v8h av = zero8;
        if (gm < M) av = *(const v8h*)(A + (size_t)gm * lda + k0 + ak);
        *(v8h*)&sA[ar][ak] = av;
        // stage W tile transposed: one b128 load + 8 b16 LDS stores per thread
        v8h wv = *(const v8h*)(W + (size_t)(k0 + wk) * ldw + bn + wc);
        #pragma unroll
        for (int i = 0; i < 8; ++i) sBt[wc + i][wk] = wv[i];
        // prefetch next K tiles
        if (ks + 1 < ksteps) {
            __builtin_prefetch(A + (size_t)gm * lda + k0 + 32 + ak, 0, 0);
            __builtin_prefetch(W + (size_t)(k0 + 32 + wk) * ldw + bn + wc, 0, 0);
        }
        __syncthreads();

        // A fragment: two contiguous 16B runs -> ds_load_b128 x2
        v8h a_lo = *(const v8h*)&sA[mrow][ahalf];
        v8h a_hi = *(const v8h*)&sA[mrow][ahalf + 16];
        v16h a = __builtin_shufflevector(a_lo, a_hi,
                 0,1,2,3,4,5,6,7,8,9,10,11,12,13,14,15);
        // B fragments: two 8-half runs (k and k+16) per lane -> ds_load_b128 x2 each
        const _Float16* bp0 = &sBt[wn0 + nidx][bhalf];
        v8h b0l = *(const v8h*)bp0;
        v8h b0h = *(const v8h*)(bp0 + 16);
        v16h b0 = __builtin_shufflevector(b0l, b0h,
                  0,1,2,3,4,5,6,7,8,9,10,11,12,13,14,15);
        const _Float16* bp1 = &sBt[wn0 + 16 + nidx][bhalf];
        v8h b1l = *(const v8h*)bp1;
        v8h b1h = *(const v8h*)(bp1 + 16);
        v16h b1 = __builtin_shufflevector(b1l, b1h,
                  0,1,2,3,4,5,6,7,8,9,10,11,12,13,14,15);

        acc0 = __builtin_amdgcn_wmma_f32_16x16x32_f16(
            false, a, false, b0, (short)0, acc0, false, false);
        acc1 = __builtin_amdgcn_wmma_f32_16x16x32_f16(
            false, a, false, b1, (short)0, acc1, false, false);
        __syncthreads();
    }

    // epilogue: D layout — M = v + 8*(lane>=16), N = lane%16
    const int rbase = bm + wm + ((lane >> 4) << 3);
    const int col0  = bn + wn0 + nidx;
    const int col1  = col0 + 16;
    const float bv0 = bias ? bias[col0] : 0.0f;
    const float bv1 = bias ? bias[col1] : 0.0f;
    #pragma unroll
    for (int v = 0; v < 8; ++v) {
        int r = rbase + v;
        if (r < M) {
            float v0 = acc0[v] + bv0;
            float v1 = acc1[v] + bv1;
            if (relu) { v0 = fmaxf(v0, 0.0f); v1 = fmaxf(v1, 0.0f); }
            Y[(size_t)r * ldy + col0] = v0;
            Y[(size_t)r * ldy + col1] = v1;
        }
    }
}

// ---------------------------------------------------------------------------
// bcomb[c] = sum_k edge_b[k] * We[k,c]      (one block of 256 threads)
// ---------------------------------------------------------------------------
__global__ void bcomb_kernel(const float* __restrict__ edge_b,
                             const float* __restrict__ We,
                             float* __restrict__ bcomb)
{
    int c = threadIdx.x;
    float acc = 0.0f;
    for (int k = 0; k < GD; ++k) acc += edge_b[k] * We[k * GD + c];
    bcomb[c] = acc;
}

__global__ void zero_f32(float* __restrict__ p, size_t n)
{
    size_t i = (size_t)blockIdx.x * blockDim.x + threadIdx.x;
    if (i < n) p[i] = 0.0f;
}

// ---------------------------------------------------------------------------
// wave per edge: raw attention logits + segment max (encoded atomicMax)
// ---------------------------------------------------------------------------
__global__ __launch_bounds__(256) void edge_alpha_kernel(
    const float* __restrict__ xl, const float* __restrict__ xr,
    const float* __restrict__ ee,
    const int* __restrict__ src, const int* __restrict__ dst,
    const float* __restrict__ att,
    float* __restrict__ alpha, unsigned* __restrict__ amax)
{
    int wid  = (int)(((size_t)blockIdx.x * blockDim.x + threadIdx.x) >> 5);
    int lane = threadIdx.x & 31;
    if (wid >= GE) return;
    int s = src[wid], d = dst[wid];
    const float* pl = xl + (size_t)s   * GD;
    const float* pr = xr + (size_t)d   * GD;
    const float* pe = ee + (size_t)wid * GD;

    float ah[GH];
    #pragma unroll
    for (int h = 0; h < GH; ++h) {
        float part = 0.0f;
        #pragma unroll
        for (int half = 0; half < 2; ++half) {
            int c = h * GC + half * 32 + lane;
            float m = pl[c] + pr[c] + pe[c];
            m = (m > 0.0f) ? m : m * NEG_SLOPE;
            part += m * att[c];
        }
        #pragma unroll
        for (int off = 16; off > 0; off >>= 1)
            part += __shfl_xor(part, off, 32);
        ah[h] = part;
    }
    if (lane == 0) {
        #pragma unroll
        for (int h = 0; h < GH; ++h) {
            alpha[(size_t)wid * GH + h] = ah[h];
            atomicMax(&amax[(size_t)d * GH + h], fenc(ah[h]));
        }
    }
}

// ---------------------------------------------------------------------------
// ex = exp(alpha - amax[dst]) in place; denom[dst] += ex
// ---------------------------------------------------------------------------
__global__ void edge_exp_kernel(const int* __restrict__ dst,
                                float* __restrict__ alpha,
                                const unsigned* __restrict__ amax,
                                float* __restrict__ denom)
{
    size_t i = (size_t)blockIdx.x * blockDim.x + threadIdx.x;
    if (i >= (size_t)GE * GH) return;
    int e = (int)(i >> 2), h = (int)(i & 3);
    int d = dst[e];
    float am = fdec(amax[(size_t)d * GH + h]);
    float ex = expf(alpha[i] - am);
    alpha[i] = ex;
    atomicAdd(&denom[(size_t)d * GH + h], ex);
}

// ---------------------------------------------------------------------------
// wave per edge: hnext[dst] += xl[src] * (ex / denom[dst])  (per head)
// ---------------------------------------------------------------------------
__global__ __launch_bounds__(256) void edge_scatter_kernel(
    const float* __restrict__ xl,
    const int* __restrict__ src, const int* __restrict__ dst,
    const float* __restrict__ alpha, const float* __restrict__ denom,
    float* __restrict__ hnext)
{
    int wid  = (int)(((size_t)blockIdx.x * blockDim.x + threadIdx.x) >> 5);
    int lane = threadIdx.x & 31;
    if (wid >= GE) return;
    int s = src[wid], d = dst[wid];
    const float* pl = xl + (size_t)s * GD;
    float* po = hnext + (size_t)d * GD;
    #pragma unroll
    for (int it = 0; it < 8; ++it) {
        int h = it >> 1;
        int c = h * GC + (it & 1) * 32 + lane;
        float a = alpha[(size_t)wid * GH + h] / denom[(size_t)d * GH + h];
        atomicAdd(&po[c], pl[c] * a);
    }
}

// ---------------------------------------------------------------------------
// add bias in place + accumulate per-channel sum / sumsq  (thread = channel)
// ---------------------------------------------------------------------------
__global__ void bias_stats_kernel(float* __restrict__ h,
                                  const float* __restrict__ bias,
                                  float* __restrict__ stats,
                                  int rows_per_block)
{
    int c  = threadIdx.x;                 // 0..255
    int r0 = blockIdx.x * rows_per_block;
    int r1 = r0 + rows_per_block; if (r1 > GN) r1 = GN;
    float b = bias[c], s = 0.0f, s2 = 0.0f;
    for (int r = r0; r < r1; ++r) {
        float v = h[(size_t)r * GD + c] + b;
        h[(size_t)r * GD + c] = v;
        s += v; s2 += v * v;
    }
    atomicAdd(&stats[c], s);
    atomicAdd(&stats[GD + c], s2);
}

// ---------------------------------------------------------------------------
// batch-norm (training-mode stats) + relu, emits f16 GEMM operand directly
// ---------------------------------------------------------------------------
__global__ void bn_relu_kernel(const float* __restrict__ in,
                               const float* __restrict__ stats,
                               const float* __restrict__ gamma,
                               const float* __restrict__ beta,
                               _Float16* __restrict__ out)
{
    size_t i = (size_t)blockIdx.x * blockDim.x + threadIdx.x;
    if (i >= (size_t)GN * GD) return;
    int c = (int)(i & (GD - 1));
    const float invN = 1.0f / (float)GN;
    float mu  = stats[c] * invN;
    float var = stats[GD + c] * invN - mu * mu;
    float v = (in[i] - mu) * rsqrtf(var + BN_EPS) * gamma[c] + beta[c];
    out[i] = (_Float16)fmaxf(v, 0.0f);
}

// ---------------------------------------------------------------------------
// pooling + output layer
// ---------------------------------------------------------------------------
__global__ void pool_cnt_kernel(const int* __restrict__ batch, float* __restrict__ cnt)
{
    int i = blockIdx.x * blockDim.x + threadIdx.x;
    if (i < GN) atomicAdd(&cnt[batch[i]], 1.0f);
}

__global__ void pool_sum_kernel(const float* __restrict__ hm,
                                const int* __restrict__ batch,
                                float* __restrict__ pooled)
{
    size_t i = (size_t)blockIdx.x * blockDim.x + threadIdx.x;
    if (i >= (size_t)GN * GC) return;
    int n = (int)(i >> 6), c = (int)(i & 63);
    atomicAdd(&pooled[batch[n] * GC + c], hm[i]);
}

__global__ void final_kernel(const float* __restrict__ pooled,
                             const float* __restrict__ cnt,
                             const float* __restrict__ out_W,
                             const float* __restrict__ out_b,
                             float* __restrict__ out)
{
    int i = blockIdx.x * blockDim.x + threadIdx.x;
    if (i >= GB * 5) return;
    int b = i / 5, k = i % 5;
    float inv = 1.0f / fmaxf(cnt[b], 1.0f);
    float acc = out_b[k];
    for (int j = 0; j < GC; ++j)
        acc += pooled[b * GC + j] * inv * out_W[j * 5 + k];
    out[i] = acc;
}

// ---------------------------------------------------------------------------
// host side
// ---------------------------------------------------------------------------
static inline void launch_zero(float* p, size_t n, hipStream_t s) {
    zero_f32<<<(int)((n + 255) / 256), 256, 0, s>>>(p, n);
}
static inline void launch_cvt(const float* src, _Float16* dst, int srows, int scols,
                              int drows, int dcols_lg2, hipStream_t s) {
    size_t n = (size_t)drows << dcols_lg2;
    cvt_f16_pad<<<(int)((n + 255) / 256), 256, 0, s>>>(src, dst, srows, scols, drows, dcols_lg2);
}

extern "C" void kernel_launch(void* const* d_in, const int* in_sizes, int n_in,
                              void* d_out, int out_size, void* d_ws, size_t ws_size,
                              hipStream_t stream) {
    (void)in_sizes; (void)n_in; (void)out_size; (void)ws_size;
    const float* x       = (const float*)d_in[0];
    const int*   eidx    = (const int*)  d_in[1];
    const float* eattr   = (const float*)d_in[2];
    const int*   batch   = (const int*)  d_in[3];
    const float* node_W  = (const float*)d_in[4];
    const float* node_b  = (const float*)d_in[5];
    const float* edge_W  = (const float*)d_in[6];
    const float* edge_b  = (const float*)d_in[7];
    const float* mlp_W   = (const float*)d_in[26];
    const float* mlp_b   = (const float*)d_in[27];
    const float* out_W   = (const float*)d_in[28];
    const float* out_b   = (const float*)d_in[29];
    const int* src = eidx;
    const int* dst = eidx + GE;

    // workspace layout (byte cursor, 256B aligned)
    char* base = (char*)d_ws;
    size_t off = 0;
    auto allocb = [&](size_t bytes) { char* p = base + off; off += (bytes + 255) & ~(size_t)255; return p; };
    auto allocf = [&](size_t n) { return (float*)allocb(n * 4); };
    auto alloch = [&](size_t n) { return (_Float16*)allocb(n * 2); };

    float*    h      = allocf((size_t)GN * GD);
    float*    hnext  = allocf((size_t)GN * GD);
    float*    xl     = allocf((size_t)GN * GD);
    float*    xr     = allocf((size_t)GN * GD);
    float*    ee     = allocf((size_t)GE * GD);      // 327 MB
    float*    alpha  = allocf((size_t)GE * GH);
    unsigned* amax   = (unsigned*)allocf((size_t)GN * GH);
    float*    denom  = allocf((size_t)GN * GH);
    float*    Wcomb  = allocf((size_t)64 * GD);
    float*    bcomb  = allocf((size_t)GD);
    float*    stats  = allocf((size_t)2 * GD);
    float*    hm     = allocf((size_t)GN * GC);
    float*    pooled = allocf((size_t)GB * GC);
    float*    cnt    = allocf((size_t)GB);
    // f16 operands (K dims padded to multiples of 32)
    _Float16* xh     = alloch((size_t)GN * 32);      // x  [N,5]  -> [N,32]
    _Float16* eh     = alloch((size_t)GE * 64);      // ea [E,63] -> [E,64]
    _Float16* hh     = alloch((size_t)GN * GD);      // node features, f16
    _Float16* nWh    = alloch((size_t)32 * GD);      // node_W [5,256] -> [32,256]
    _Float16* eWh    = alloch((size_t)63 * GD);      // edge_W f16 (K=256 ok)
    _Float16* Wlh    = alloch((size_t)GD * GD);
    _Float16* Wrh    = alloch((size_t)GD * GD);
    _Float16* Weh    = alloch((size_t)GD * GD);
    _Float16* Wcombh = alloch((size_t)64 * GD);      // [63,256] -> [64,256]
    _Float16* mWh    = alloch((size_t)GD * GC);

    dim3 blk(256);

    // one-time f16 conversions
    launch_cvt(x,      xh,  GN, 5,   GN, 5, stream);   // dcols = 32
    launch_cvt(eattr,  eh,  GE, 63,  GE, 6, stream);   // dcols = 64
    launch_cvt(node_W, nWh, 5,  GD,  32, 8, stream);   // dcols = 256
    launch_cvt(edge_W, eWh, 63, GD,  63, 8, stream);
    launch_cvt(mlp_W,  mWh, GD, GC,  GD, 6, stream);   // dcols = 64

    // node embedding: h = x @ node_W + node_b   (M=N, K=32 padded)
    {
        dim3 grd((GN + 63) / 64, GD / 64);
        wmma_gemm_f16<<<grd, blk, 0, stream>>>(xh, 32, nWh, GD, node_b, h, GD, GN, 32, 0);
    }
    launch_cvt(h, hh, GN, GD, GN, 8, stream);

    for (int l = 0; l < 2; ++l) {
        const float* Wl  = (const float*)d_in[8 + 9 * l + 0];
        const float* bl  = (const float*)d_in[8 + 9 * l + 1];
        const float* Wr  = (const float*)d_in[8 + 9 * l + 2];
        const float* br  = (const float*)d_in[8 + 9 * l + 3];
        const float* We  = (const float*)d_in[8 + 9 * l + 4];
        const float* att = (const float*)d_in[8 + 9 * l + 5];
        const float* bias= (const float*)d_in[8 + 9 * l + 6];
        const float* gam = (const float*)d_in[8 + 9 * l + 7];
        const float* bet = (const float*)d_in[8 + 9 * l + 8];

        launch_cvt(Wl, Wlh, GD, GD, GD, 8, stream);
        launch_cvt(Wr, Wrh, GD, GD, GD, 8, stream);
        launch_cvt(We, Weh, GD, GD, GD, 8, stream);

        // folded edge path: Wcomb = edge_W @ We ; bcomb = edge_b @ We
        {
            dim3 grd(1, GD / 64);
            wmma_gemm_f16<<<grd, blk, 0, stream>>>(eWh, GD, Weh, GD, nullptr, Wcomb, GD, 63, GD, 0);
        }
        launch_cvt(Wcomb, Wcombh, 63, GD, 64, 8, stream);
        bcomb_kernel<<<1, GD, 0, stream>>>(edge_b, We, bcomb);

        // ee = edge_attr @ Wcomb + bcomb   (M=E, K=64 padded)
        {
            dim3 grd((GE + 63) / 64, GD / 64);
            wmma_gemm_f16<<<grd, blk, 0, stream>>>(eh, 64, Wcombh, GD, bcomb, ee, GD, GE, 64, 0);
        }
        // xl = h @ Wl + bl ; xr = h @ Wr + br
        {
            dim3 grd((GN + 63) / 64, GD / 64);
            wmma_gemm_f16<<<grd, blk, 0, stream>>>(hh, GD, Wlh, GD, bl, xl, GD, GN, GD, 0);
            wmma_gemm_f16<<<grd, blk, 0, stream>>>(hh, GD, Wrh, GD, br, xr, GD, GN, GD, 0);
        }

        // attention softmax + scatter
        launch_zero(hnext, (size_t)GN * GD, stream);
        launch_zero(denom, (size_t)GN * GH, stream);
        launch_zero((float*)amax, (size_t)GN * GH, stream);   // 0 == encoded -inf

        edge_alpha_kernel<<<GE / 8, 256, 0, stream>>>(xl, xr, ee, src, dst, att, alpha, amax);
        edge_exp_kernel<<<(GE * GH + 255) / 256, 256, 0, stream>>>(dst, alpha, amax, denom);
        edge_scatter_kernel<<<GE / 8, 256, 0, stream>>>(xl, src, dst, alpha, denom, hnext);

        // bias + batch-norm stats + bn/relu -> f16 features for next GEMMs
        launch_zero(stats, 2 * GD, stream);
        bias_stats_kernel<<<(GN + 79) / 80, GD, 0, stream>>>(hnext, bias, stats, 80);
        bn_relu_kernel<<<(int)(((size_t)GN * GD + 255) / 256), 256, 0, stream>>>(hnext, stats, gam, bet, hh);
    }

    // hm = relu(h @ mlp_W + mlp_b)  (M=N, K=256, Nw=64)
    {
        dim3 grd((GN + 63) / 64, 1);
        wmma_gemm_f16<<<grd, blk, 0, stream>>>(hh, GD, mWh, GC, mlp_b, hm, GC, GN, GD, 1);
    }

    // global mean pool + output layer
    launch_zero(pooled, (size_t)GB * GC, stream);
    launch_zero(cnt, GB, stream);
    pool_cnt_kernel<<<(GN + 255) / 256, 256, 0, stream>>>(batch, cnt);
    pool_sum_kernel<<<(int)(((size_t)GN * GC + 255) / 256), 256, 0, stream>>>(hm, batch, pooled);
    final_kernel<<<2, 256, 0, stream>>>(pooled, cnt, out_W, out_b, (float*)d_out);
}
